// RankingLossFunc_24575802868425
// MI455X (gfx1250) — matrix operand order repaired and checked by reading, not compiled, not run on hardware
//
#include <hip/hip_runtime.h>
#include <hip/hip_bf16.h>
#include <math.h>

// Problem constants (from reference): X, Y are [8192, 1024] float32.
#define NROWS 8192
#define DIM   1024
#define DELTA 0.1f
#define EPS_N 1e-8f

typedef __attribute__((ext_vector_type(16))) __bf16 v16bf;
typedef __attribute__((ext_vector_type(8)))  __bf16 v8bf;
typedef __attribute__((ext_vector_type(8)))  float  v8f;

// ---------------------------------------------------------------------------
// Kernel 1: per-row normalization + bf16 hi/lo split + exact diagonal pos[i].
//   Xn = X / max(||X||, eps); stored as Xhi + Xlo (bf16 pair ~ f32 accuracy)
//   pos[i] = (X_i . Y_i) * rxinv * ryinv  computed fully in f32.
// One 256-thread block per row; each thread handles 4 of the 1024 elements.
// ---------------------------------------------------------------------------
__global__ void __launch_bounds__(256)
rownorm_kernel(const float* __restrict__ X, const float* __restrict__ Y,
               __bf16* __restrict__ Xhi, __bf16* __restrict__ Xlo,
               __bf16* __restrict__ Yhi, __bf16* __restrict__ Ylo,
               float* __restrict__ pos)
{
    __shared__ float sX[256], sY[256], sXY[256];
    __shared__ float bcast[2];
    const int i = blockIdx.x;
    const int t = threadIdx.x;
    const float* xr = X + (size_t)i * DIM;
    const float* yr = Y + (size_t)i * DIM;

    float xv[4], yv[4];
    float sx = 0.f, sy = 0.f, sxy = 0.f;
#pragma unroll
    for (int q = 0; q < 4; ++q) {
        const int d = t + q * 256;
        const float x = xr[d], y = yr[d];
        xv[q] = x; yv[q] = y;
        sx += x * x; sy += y * y; sxy += x * y;
    }
    sX[t] = sx; sY[t] = sy; sXY[t] = sxy;
    __syncthreads();
    for (int off = 128; off > 0; off >>= 1) {
        if (t < off) { sX[t] += sX[t+off]; sY[t] += sY[t+off]; sXY[t] += sXY[t+off]; }
        __syncthreads();
    }
    if (t == 0) {
        const float rxi = 1.0f / fmaxf(sqrtf(sX[0]), EPS_N);
        const float ryi = 1.0f / fmaxf(sqrtf(sY[0]), EPS_N);
        bcast[0] = rxi; bcast[1] = ryi;
        pos[i] = sXY[0] * rxi * ryi;   // exact f32 diagonal of S
    }
    __syncthreads();
    const float rxi = bcast[0], ryi = bcast[1];
#pragma unroll
    for (int q = 0; q < 4; ++q) {
        const size_t d = (size_t)i * DIM + (t + q * 256);
        const float xn = xv[q] * rxi;
        const __bf16 xh = (__bf16)xn;                 // RNE
        const __bf16 xl = (__bf16)(xn - (float)xh);   // residual
        Xhi[d] = xh; Xlo[d] = xl;
        const float yn = yv[q] * ryi;
        const __bf16 yh = (__bf16)yn;
        const __bf16 yl = (__bf16)(yn - (float)yh);
        Yhi[d] = yh; Ylo[d] = yl;
    }
}

// ---------------------------------------------------------------------------
// A-fragment loader for V_WMMA_F32_16X16X32_BF16 (16-bit A 16x32, contiguous K order used by this code, same as B):
//   lane L (0..15):  M = L,     elems 0..15 -> K=0..15
//   lane L (16..31): M = L-16,  elems 0..15 -> K=16..31
// Row-major bf16 storage => two contiguous 16-byte loads per lane.
// ---------------------------------------------------------------------------
__device__ __forceinline__ v16bf load_a_frag(const __bf16* __restrict__ p)
{
    const v8bf a0 = *(const v8bf*)(p);        // K + khalf + [0..7]
    const v8bf a1 = *(const v8bf*)(p + 8);    // K + khalf + [8..15]
    return __builtin_shufflevector(a0, a1, 0,1,2,3,4,5,6,7,8,9,10,11,12,13,14,15);
}

// ---------------------------------------------------------------------------
// Kernel 2: fused S-tile GEMM (bf16 hi/lo split -> ~f32 accuracy) + hinge +
// diagonal mask + block reduction.
//   Block tile 128(M) x 128(N); 8 waves arranged 4(M) x 2(N); per-wave tile
//   32 x 64 = 2 x 4 WMMA subtiles. K loop: 1024/32 = 32 steps.
//   Per wave per K-step: 24 v_wmma vs 24 global_load_b128 (1:1), with each
//   B fragment amortized over 2 A rows and each A fragment over 4 B columns.
// B-fragment (32x16 K-major): lane L: N = L%16, elems e -> K = 16*(L/16)+e
// => one contiguous 32-byte load per lane from row (column of S) of Y.
// All operand arrays (64 MB total) are L2-resident (192 MB L2).
// ---------------------------------------------------------------------------
__global__ void __launch_bounds__(256)
gemm_hinge_kernel(const __bf16* __restrict__ Xhi, const __bf16* __restrict__ Xlo,
                  const __bf16* __restrict__ Yhi, const __bf16* __restrict__ Ylo,
                  const float* __restrict__ pos, float* __restrict__ partials)
{
    const int lane = threadIdx.x & 31;
    const int wid  = threadIdx.x >> 5;
    const int wm   = wid & 3;          // 4 waves along M
    const int wn   = wid >> 2;         // 2 waves along N
    const int rowBase = blockIdx.y * 128 + wm * 32;   // 32 rows per wave
    const int colBase = blockIdx.x * 128 + wn * 64;   // 64 cols per wave
    const int l16 = lane & 15;
    const int hf  = lane >> 4;         // lane half selects K sub-group

    v8f acc[2][4];
#pragma unroll
    for (int mi = 0; mi < 2; ++mi)
#pragma unroll
        for (int ni = 0; ni < 4; ++ni) acc[mi][ni] = (v8f){};

    const int aK = hf * 16;            // A fragment K base within 32-chunk
    const int bK = hf * 16;            // B fragment K base within 32-chunk
    size_t ar[2], br[4];
#pragma unroll
    for (int mi = 0; mi < 2; ++mi)
        ar[mi] = (size_t)(rowBase + mi * 16 + l16) * DIM + aK;
#pragma unroll
    for (int ni = 0; ni < 4; ++ni)
        br[ni] = (size_t)(colBase + ni * 16 + l16) * DIM + bK;

    for (int k0 = 0; k0 < DIM; k0 += 32) {
        // Prefetch next K-chunk toward L0 (gfx1250 global_prefetch_b8).
        if (k0 + 32 < DIM) {
            __builtin_prefetch(Xhi + ar[0] + k0 + 32, 0, 3);
            __builtin_prefetch(Yhi + br[0] + k0 + 32, 0, 3);
        }
        v16bf ah[2], al[2], bh[4], bl[4];
#pragma unroll
        for (int mi = 0; mi < 2; ++mi) {
            ah[mi] = load_a_frag(Xhi + ar[mi] + k0);
            al[mi] = load_a_frag(Xlo + ar[mi] + k0);
        }
#pragma unroll
        for (int ni = 0; ni < 4; ++ni) {
            bh[ni] = *(const v16bf*)(Yhi + br[ni] + k0);
            bl[ni] = *(const v16bf*)(Ylo + br[ni] + k0);
        }
#pragma unroll
        for (int mi = 0; mi < 2; ++mi) {
#pragma unroll
            for (int ni = 0; ni < 4; ++ni) {
                // hi*hi + hi*lo + lo*hi  (drop lo*lo: ~2^-32 relative)
                acc[mi][ni] = __builtin_amdgcn_wmma_f32_16x16x32_bf16(
                    false, ah[mi], false, bh[ni], (short)0, acc[mi][ni], false, false);
                acc[mi][ni] = __builtin_amdgcn_wmma_f32_16x16x32_bf16(
                    false, ah[mi], false, bl[ni], (short)0, acc[mi][ni], false, false);
                acc[mi][ni] = __builtin_amdgcn_wmma_f32_16x16x32_bf16(
                    false, al[mi], false, bh[ni], (short)0, acc[mi][ni], false, false);
            }
        }
    }

    // Epilogue: C/D layout (ISA 7.12.2): VGPR v, lanes 0-15 -> M=v, lanes
    // 16-31 -> M=v+8; N = lane%16. Fused hinge + diagonal mask + lane sum.
    float lsum = 0.f;
#pragma unroll
    for (int mi = 0; mi < 2; ++mi) {
        const int ibase = rowBase + mi * 16 + hf * 8;
#pragma unroll
        for (int ni = 0; ni < 4; ++ni) {
            const int j = colBase + ni * 16 + l16;
#pragma unroll
            for (int v = 0; v < 8; ++v) {
                const int i = ibase + v;
                const float h = fmaxf(0.f, DELTA - pos[i] + acc[mi][ni][v]);
                lsum += (i == j) ? 0.f : h;
            }
        }
    }

    __shared__ float red[256];
    red[threadIdx.x] = lsum;
    __syncthreads();
    for (int off = 128; off > 0; off >>= 1) {
        if (threadIdx.x < off) red[threadIdx.x] += red[threadIdx.x + off];
        __syncthreads();
    }
    if (threadIdx.x == 0)
        partials[blockIdx.y * gridDim.x + blockIdx.x] = red[0];
}

// ---------------------------------------------------------------------------
// Kernel 3: deterministic final reduction (fixed summation order, no float
// atomics -> bit-identical across graph replays).
// ---------------------------------------------------------------------------
__global__ void __launch_bounds__(256)
final_reduce_kernel(const float* __restrict__ partials, int n, float* __restrict__ out)
{
    __shared__ float red[256];
    const int t = threadIdx.x;
    float s = 0.f;
    for (int idx = t; idx < n; idx += 256) s += partials[idx];
    red[t] = s;
    __syncthreads();
    for (int off = 128; off > 0; off >>= 1) {
        if (t < off) red[t] += red[t + off];
        __syncthreads();
    }
    if (t == 0) out[0] = red[0];
}

extern "C" void kernel_launch(void* const* d_in, const int* in_sizes, int n_in,
                              void* d_out, int out_size, void* d_ws, size_t ws_size,
                              hipStream_t stream)
{
    const float* X = (const float*)d_in[0];
    const float* Y = (const float*)d_in[1];

    // Workspace layout (~64.05 MB): 4 bf16 matrices + pos + block partials.
    char* ws = (char*)d_ws;
    const size_t matBytes = (size_t)NROWS * DIM * sizeof(__bf16);   // 16 MB
    __bf16* Xhi = (__bf16*)(ws + 0 * matBytes);
    __bf16* Xlo = (__bf16*)(ws + 1 * matBytes);
    __bf16* Yhi = (__bf16*)(ws + 2 * matBytes);
    __bf16* Ylo = (__bf16*)(ws + 3 * matBytes);
    float*  pos = (float*)(ws + 4 * matBytes);
    float*  partials = pos + NROWS;

    rownorm_kernel<<<NROWS, 256, 0, stream>>>(X, Y, Xhi, Xlo, Yhi, Ylo, pos);

    dim3 grid(NROWS / 128, NROWS / 128);   // 64 x 64 = 4096 tiles of 128 x 128
    gemm_hinge_kernel<<<grid, 256, 0, stream>>>(Xhi, Xlo, Yhi, Ylo, pos, partials);

    const int nPart = (NROWS / 128) * (NROWS / 128);
    final_reduce_kernel<<<1, 256, 0, stream>>>(partials, nPart, (float*)d_out);
}
